// ResidualVQ_37598143709959
// MI455X (gfx1250) — compile-verified, matrix operand order checked by reading, not run on hardware
//
#include <hip/hip_runtime.h>
#include <stdint.h>

typedef _Float16 v8h  __attribute__((ext_vector_type(8)));
typedef _Float16 v16h __attribute__((ext_vector_type(16)));
typedef float    v8f  __attribute__((ext_vector_type(8)));

#define DDIM    256
#define NCODE   1024
#define NQ      4
#define TOKBLK  128
#define CHUNK   64
#define CBPITCH 264                       // halves: 256 + 8 pad -> 528B rows, bank-skewed
#define LCB_BYTES (CHUNK * CBPITCH * 2)   // 33792 per buffer
#define HN_OFF    (2 * LCB_BYTES)         // 67584
#define RES_OFF   (HN_OFF + 2 * 256)      // 68096 (16B aligned)
#define RES_PITCH 528                     // bytes per thread (512B data + skew pad)
#define SMEM_TOTAL (RES_OFF + 256 * RES_PITCH)   // 203264 bytes < 320KB

// ---- CDNA5 async global->LDS (ASYNCcnt-tracked, bypasses VGPRs) ----
__device__ __forceinline__ unsigned ldsoff(const void* p) {
  // generic LDS pointer: low 32 bits are the wave-relative LDS byte offset
  return (unsigned)(uintptr_t)p;
}
__device__ __forceinline__ void async_b128(unsigned l, const void* g) {
  asm volatile("global_load_async_to_lds_b128 %0, %1, off"
               :: "v"(l), "v"(g) : "memory");
}
__device__ __forceinline__ void async_b32(unsigned l, const void* g) {
  asm volatile("global_load_async_to_lds_b32 %0, %1, off"
               :: "v"(l), "v"(g) : "memory");
}

// ---------------- prep: f32 codebook -> f16 copy + 0.5*||e||^2 ----------------
__global__ __launch_bounds__(256) void rvq_prep(const float* __restrict__ cb,
                                                _Float16* __restrict__ cbh,
                                                float* __restrict__ hn) {
  __shared__ float red[256];
  const int row = blockIdx.x;          // q*NCODE + c
  const int t = threadIdx.x;           // dim
  float v = cb[(size_t)row * DDIM + t];
  cbh[(size_t)row * DDIM + t] = (_Float16)v;
  red[t] = v * v;
  __syncthreads();
  for (int s = 128; s > 0; s >>= 1) {
    if (t < s) red[t] += red[t + s];
    __syncthreads();
  }
  if (t == 0) hn[row] = 0.5f * red[0];
}

// issue one 64-code chunk (32KB + 64 norms) as async-to-LDS; 9 asyncs per wave
__device__ __forceinline__ void issue_chunk(char* smem, const _Float16* cq,
                                            const float* hq, int cb0, int buf,
                                            int tid) {
  const int row = tid >> 2;                 // 0..63
  const int col = (tid & 3) * 64;           // halves
  const _Float16* g = cq + (size_t)(cb0 + row) * DDIM + col;
  char* l = smem + buf * LCB_BYTES + row * (CBPITCH * 2) + col * 2;
#pragma unroll
  for (int i = 0; i < 8; ++i)
    async_b128(ldsoff(l + i * 16), g + i * 8);
  if ((tid & 3) == 0)   // 8 active lanes in EVERY wave -> uniform 9 asyncs/wave
    async_b32(ldsoff(smem + HN_OFF + buf * 256 + row * 4), hq + cb0 + row);
}

// ---------------- main: residual VQ, 4 layers fused per token tile ------------
__global__ __launch_bounds__(256, 1) void rvq_main(
    const float* __restrict__ x, const float* __restrict__ cbf,
    const _Float16* __restrict__ cbh, const float* __restrict__ hn,
    float* __restrict__ outq, int* __restrict__ outidx) {
  extern __shared__ __align__(16) char smem[];

  const int tid  = threadIdx.x;
  const int lane = tid & 31;
  const int wv   = tid >> 5;          // wave 0..7, owns 16 tokens
  const int trow = lane & 15;         // token-in-tile; also code row for A frags
  const int off0 = (lane >> 4) * 8;   // K-offset split between lane halves
  const size_t tokn = (size_t)blockIdx.x * TOKBLK + wv * 16 + trow;

  const float* xr = x + tokn * DDIM;
  // per-lane private residual block in LDS, fragment-ordered:
  // myres[b*16+j] <-> dim b*32 + off0 + j + (j&8)
  float* myres = (float*)(smem + RES_OFF + tid * RES_PITCH);

#pragma unroll
  for (int b = 0; b < 8; ++b) {
    const float* p = xr + b * 32 + off0;
    float4 a0 = *(const float4*)(p + 0);
    float4 a1 = *(const float4*)(p + 4);
    float4 a2 = *(const float4*)(p + 16);
    float4 a3 = *(const float4*)(p + 20);
    *(float4*)(myres + b * 16 + 0)  = a0;
    *(float4*)(myres + b * 16 + 4)  = a1;
    *(float4*)(myres + b * 16 + 8)  = a2;
    *(float4*)(myres + b * 16 + 12) = a3;
  }

#pragma unroll 1
  for (int q = 0; q < NQ; ++q) {
    const _Float16* cq = cbh + (size_t)q * NCODE * DDIM;
    const float*    hq = hn + q * NCODE;

    // prefetch chunk 0 into buffer 0 (overlaps with the f16 conversion below)
    issue_chunk(smem, cq, hq, 0, 0, tid);

    // B fragments: this wave's 16 tokens, K=256, f16, once per layer
    v16h bf[8];
#pragma unroll
    for (int b = 0; b < 8; ++b) {
      float4 r0 = *(const float4*)(myres + b * 16 + 0);
      float4 r1 = *(const float4*)(myres + b * 16 + 4);
      float4 r2 = *(const float4*)(myres + b * 16 + 8);
      float4 r3 = *(const float4*)(myres + b * 16 + 12);
      bf[b][0]=(_Float16)r0.x;  bf[b][1]=(_Float16)r0.y;
      bf[b][2]=(_Float16)r0.z;  bf[b][3]=(_Float16)r0.w;
      bf[b][4]=(_Float16)r1.x;  bf[b][5]=(_Float16)r1.y;
      bf[b][6]=(_Float16)r1.z;  bf[b][7]=(_Float16)r1.w;
      bf[b][8]=(_Float16)r2.x;  bf[b][9]=(_Float16)r2.y;
      bf[b][10]=(_Float16)r2.z; bf[b][11]=(_Float16)r2.w;
      bf[b][12]=(_Float16)r3.x; bf[b][13]=(_Float16)r3.y;
      bf[b][14]=(_Float16)r3.z; bf[b][15]=(_Float16)r3.w;
    }

    float bestv = 3.0e38f;
    int   besti = 0;

#pragma unroll 1
    for (int c = 0; c < NCODE / CHUNK; ++c) {
      const int cb0 = c * CHUNK;
      const int cur = c & 1;
      if (c + 1 < NCODE / CHUNK) {
        issue_chunk(smem, cq, hq, cb0 + CHUNK, cur ^ 1, tid);
        // in-order completion: <=9 outstanding => chunk c fully landed
        asm volatile("s_wait_asynccnt 0x9" ::: "memory");
      } else {
        asm volatile("s_wait_asynccnt 0x0" ::: "memory");
      }
      __syncthreads();   // chunk c visible to all waves

      const _Float16* lcb = (const _Float16*)(smem + cur * LCB_BYTES);
      const float*    lhn = (const float*)(smem + HN_OFF + cur * 256);

      // 4 independent accumulator chains (one per 16-code tile)
      v8f acc[4];
#pragma unroll
      for (int ct = 0; ct < 4; ++ct) acc[ct] = (v8f){};
#pragma unroll
      for (int kb = 0; kb < 8; ++kb) {
#pragma unroll
        for (int ct = 0; ct < 4; ++ct) {
          const _Float16* ap = lcb + (ct * 16 + trow) * CBPITCH + kb * 32 + off0;
          v8h alo = *(const v8h*)ap;          // K = kb*32+off0 .. +8
          v8h ahi = *(const v8h*)(ap + 16);   // K = kb*32+off0+16 .. +24
          v16h af = __builtin_shufflevector(alo, ahi,
              0,1,2,3,4,5,6,7,8,9,10,11,12,13,14,15);
          acc[ct] = __builtin_amdgcn_wmma_f32_16x16x32_f16(
              false, af, false, bf[kb], (short)0, acc[ct], false, false);
        }
      }
      // argmin of 0.5||e||^2 - r.e (|| r ||^2 constant per token)
#pragma unroll
      for (int ct = 0; ct < 4; ++ct) {
#pragma unroll
        for (int v = 0; v < 8; ++v) {
          const float val = lhn[ct * 16 + off0 + v] - acc[ct][v];
          const int  code = cb0 + ct * 16 + off0 + v;
          if (val < bestv) { bestv = val; besti = code; }
        }
      }
      __syncthreads();   // all waves done reading buf 'cur' before it refills
    }

    // merge disjoint code halves between lane L and L+16 (same token)
    const float ov = __shfl_xor(bestv, 16, 32);
    const int   oi = __shfl_xor(besti, 16, 32);
    if (ov < bestv || (ov == bestv && oi < besti)) { bestv = ov; besti = oi; }

    if (lane < 16) outidx[tokn * NQ + q] = besti;

    // exact f32 gather + residual update (output depends only on indices)
    const float* er = cbf + ((size_t)q * NCODE + besti) * DDIM;
#pragma unroll
    for (int b = 0; b < 8; ++b) {
      const float* p = er + b * 32 + off0;
      float4 e0 = *(const float4*)(p + 0);
      float4 e1 = *(const float4*)(p + 4);
      float4 e2 = *(const float4*)(p + 16);
      float4 e3 = *(const float4*)(p + 20);
      float4 r0 = *(const float4*)(myres + b * 16 + 0);
      float4 r1 = *(const float4*)(myres + b * 16 + 4);
      float4 r2 = *(const float4*)(myres + b * 16 + 8);
      float4 r3 = *(const float4*)(myres + b * 16 + 12);
      r0.x-=e0.x; r0.y-=e0.y; r0.z-=e0.z; r0.w-=e0.w;
      r1.x-=e1.x; r1.y-=e1.y; r1.z-=e1.z; r1.w-=e1.w;
      r2.x-=e2.x; r2.y-=e2.y; r2.z-=e2.z; r2.w-=e2.w;
      r3.x-=e3.x; r3.y-=e3.y; r3.z-=e3.z; r3.w-=e3.w;
      *(float4*)(myres + b * 16 + 0)  = r0;
      *(float4*)(myres + b * 16 + 4)  = r1;
      *(float4*)(myres + b * 16 + 8)  = r2;
      *(float4*)(myres + b * 16 + 12) = r3;
    }
  }

  // quant_out = x - final residual (== exact sum of gathered codebook rows)
  float* oq = outq + tokn * DDIM;
#pragma unroll
  for (int b = 0; b < 8; ++b) {
    const float* p = xr + b * 32 + off0;
    float4 a0 = *(const float4*)(p + 0);
    float4 a1 = *(const float4*)(p + 4);
    float4 a2 = *(const float4*)(p + 16);
    float4 a3 = *(const float4*)(p + 20);
    float4 r0 = *(const float4*)(myres + b * 16 + 0);
    float4 r1 = *(const float4*)(myres + b * 16 + 4);
    float4 r2 = *(const float4*)(myres + b * 16 + 8);
    float4 r3 = *(const float4*)(myres + b * 16 + 12);
    float4 o0 = make_float4(a0.x-r0.x, a0.y-r0.y, a0.z-r0.z, a0.w-r0.w);
    float4 o1 = make_float4(a1.x-r1.x, a1.y-r1.y, a1.z-r1.z, a1.w-r1.w);
    float4 o2 = make_float4(a2.x-r2.x, a2.y-r2.y, a2.z-r2.z, a2.w-r2.w);
    float4 o3 = make_float4(a3.x-r3.x, a3.y-r3.y, a3.z-r3.z, a3.w-r3.w);
    *(float4*)(oq + b * 32 + off0 + 0)  = o0;
    *(float4*)(oq + b * 32 + off0 + 4)  = o1;
    *(float4*)(oq + b * 32 + off0 + 16) = o2;
    *(float4*)(oq + b * 32 + off0 + 20) = o3;
  }
}

extern "C" void kernel_launch(void* const* d_in, const int* in_sizes, int n_in,
                              void* d_out, int out_size, void* d_ws, size_t ws_size,
                              hipStream_t stream) {
  const float* x   = (const float*)d_in[0];
  const float* cbf = (const float*)d_in[1];
  const int N = in_sizes[0] / DDIM;                 // 65536 tokens

  _Float16* cbh = (_Float16*)d_ws;                  // 2 MB f16 codebooks
  float* hnw = (float*)((char*)d_ws +
      (size_t)NQ * NCODE * DDIM * sizeof(_Float16)); // 16 KB half-norms
  float* outq = (float*)d_out;
  int* outidx = (int*)((float*)d_out + (size_t)N * DDIM);

  rvq_prep<<<NQ * NCODE, 256, 0, stream>>>(cbf, cbh, hnw);
  rvq_main<<<N / TOKBLK, 256, SMEM_TOTAL, stream>>>(x, cbf, cbh, hnw, outq, outidx);
}
